// LSTM_90323162235594
// MI455X (gfx1250) — matrix operand order repaired and checked by reading, not compiled, not run on hardware
//
#include <hip/hip_runtime.h>
#include <math.h>

typedef __bf16 bf16_t;
typedef __attribute__((ext_vector_type(16))) __bf16 v16bf;
typedef __attribute__((ext_vector_type(8)))  __bf16 v8bf;
typedef __attribute__((ext_vector_type(8)))  float  v8f;

#define T_SEQ 512
#define BATCH 64
#define DIM   1024            // D == H
#define NTOT  (4 * DIM)       // 4096 stacked gate rows
#define KSTEPS (DIM / 32)     // 32 K-iterations (even)

// ---- workspace layout (bytes), all offsets 256B aligned ----
#define BAR_OFF    0u
#define WXB_OFF    256u
#define WHB_OFF    (WXB_OFF + (size_t)NTOT * DIM * 2)       // + 8 MB
#define HBUF_OFF   (WHB_OFF + (size_t)NTOT * DIM * 2)       // + 8 MB
#define XPROJ_OFF  (HBUF_OFF + (size_t)2 * BATCH * DIM * 2) // + 256 KB
#define NBLK 32

// ---------------- WMMA fragment helpers (CDNA5 layouts) ----------------

// A fragment 16x32 bf16: lane<16 -> M=lane, K runs {0..7, 16..23};
//                        lane>=16 -> M=lane-16, K runs {8..15, 24..31}
__device__ inline v16bf load_a_bf16(const bf16_t* rowbase, int ld, int lane, int k0) {
  const bf16_t* p = rowbase + (size_t)(lane & 15) * ld + k0 + ((lane & 16) ? 8 : 0);
  v8bf lo = *(const v8bf*)(p);
  v8bf hi = *(const v8bf*)(p + 16);
  v16bf r;
#pragma unroll
  for (int i = 0; i < 8; ++i) { r[i] = lo[i]; r[i + 8] = hi[i]; }
  return r;
}

// Same A layout but source is f32, converted to bf16 in-register (RNE).
__device__ inline v16bf load_a_f32(const float* rowbase, int ld, int lane, int k0) {
  const float* p = rowbase + (size_t)(lane & 15) * ld + k0 + ((lane & 16) ? 8 : 0);
  v16bf r;
#pragma unroll
  for (int i = 0; i < 8; ++i) {
    r[i]     = (__bf16)p[i];
    r[i + 8] = (__bf16)p[i + 16];
  }
  return r;
}

// B fragment 32x16 bf16 from a row-major [N,K] weight matrix:
// lane picks column N = (lane&15); lane<16 holds K runs {0..7, 16..23}, lane>=16 K runs {8..15, 24..31}.
__device__ inline v16bf load_b_bf16(const bf16_t* nrowbase, int ld, int lane, int k0) {
  const bf16_t* p = nrowbase + (size_t)(lane & 15) * ld + k0 + ((lane & 16) ? 8 : 0);
  v8bf lo = *(const v8bf*)(p);
  v8bf hi = *(const v8bf*)(p + 16);
  v16bf r;
#pragma unroll
  for (int i = 0; i < 8; ++i) { r[i] = lo[i]; r[i + 8] = hi[i]; }
  return r;
}

__device__ inline v8f wmma_bf16(v16bf a, v16bf b, v8f c) {
  return __builtin_amdgcn_wmma_f32_16x16x32_bf16(false, a, false, b, (short)0, c,
                                                 false, false);
}

__device__ inline float sigmoidf(float x) { return 1.0f / (1.0f + expf(-x)); }

// ---------------- kernel 1: f32 -> bf16 weight conversion + barrier init ----------------
__global__ __launch_bounds__(256) void cvt_weights(const float* __restrict__ Wx,
                                                   const float* __restrict__ Wh,
                                                   bf16_t* __restrict__ WxB,
                                                   bf16_t* __restrict__ WhB,
                                                   int* __restrict__ bar) {
  if (blockIdx.x == 0 && threadIdx.x < 8) bar[threadIdx.x] = 0;
  const size_t n = (size_t)NTOT * DIM;  // elements per weight tensor
  size_t base = ((size_t)blockIdx.x * blockDim.x + threadIdx.x) * 4;
#pragma unroll
  for (int j = 0; j < 4; ++j) {
    size_t idx = base + j;
    if (idx < n)            WxB[idx]     = (__bf16)Wx[idx];
    else if (idx < 2 * n)   WhB[idx - n] = (__bf16)Wh[idx - n];
  }
}

// ---------------- kernel 2: xproj[t,g,b,h] = x @ Wx^T + bx ----------------
// M = T*B (16-row tiles), N = 4H. One wave: 1 M-tile x 4 N-tiles (A reuse).
// K-loop is software-pipelined with a manual 2x unroll and two alternating
// fragment register sets (no rotation copies).
__global__ __launch_bounds__(256) void xproj_gemm(const float* __restrict__ x,
                                                  const bf16_t* __restrict__ WxB,
                                                  const float* __restrict__ bx,
                                                  float* __restrict__ xproj) {
  const int wave = threadIdx.x >> 5;
  const int lane = threadIdx.x & 31;
  const int mt   = blockIdx.x;               // 0..2047
  const int ngrp = blockIdx.y * 8 + wave;    // 0..63 (each = 4 N-tiles)
  const int m0   = mt * 16;
  const float* arow = x + (size_t)m0 * DIM;

  const bf16_t* brow[4];
#pragma unroll
  for (int j = 0; j < 4; ++j) brow[j] = WxB + (size_t)((ngrp * 4 + j) * 16) * DIM;

  v8f acc[4];
#pragma unroll
  for (int j = 0; j < 4; ++j) {
    int n = (ngrp * 4 + j) * 16 + (lane & 15);
    float bias = bx[n];                      // bx flat [4*H] == N index
#pragma unroll
    for (int r = 0; r < 8; ++r) acc[j][r] = bias;
  }

  v16bf aA, aB, bA[4], bB[4];

  // prologue: stage set A at k=0
  aA = load_a_f32(arow, DIM, lane, 0);
#pragma unroll
  for (int j = 0; j < 4; ++j) bA[j] = load_b_bf16(brow[j], DIM, lane, 0);

  for (int kk = 0; kk < KSTEPS; kk += 2) {
    // stage set B at k+32, then consume set A
    int k1 = (kk + 1) * 32;
    aB = load_a_f32(arow, DIM, lane, k1);
#pragma unroll
    for (int j = 0; j < 4; ++j) bB[j] = load_b_bf16(brow[j], DIM, lane, k1);
#pragma unroll
    for (int j = 0; j < 4; ++j) acc[j] = wmma_bf16(aA, bA[j], acc[j]);

    // stage set A at k+64 (wraps to 0 on tail: dead but in-bounds), consume set B
    int k2 = (kk + 2 < KSTEPS) ? (kk + 2) * 32 : 0;
    aA = load_a_f32(arow, DIM, lane, k2);
#pragma unroll
    for (int j = 0; j < 4; ++j) bA[j] = load_b_bf16(brow[j], DIM, lane, k2);
#pragma unroll
    for (int j = 0; j < 4; ++j) acc[j] = wmma_bf16(aB, bB[j], acc[j]);
  }

  const int rofs = (lane & 16) ? 8 : 0;
#pragma unroll
  for (int j = 0; j < 4; ++j) {
    int n = (ngrp * 4 + j) * 16 + (lane & 15);
    int g = n >> 10;
    int h = n & (DIM - 1);
#pragma unroll
    for (int r = 0; r < 8; ++r) {
      int m = m0 + r + rofs;                 // m = t*64 + b ; 16-row tile stays in one t
      int t = m >> 6;
      int b = m & 63;
      xproj[(((size_t)t * 4 + g) * BATCH + b) * DIM + h] = acc[j][r];
    }
  }
}

// ---------------- device-wide barrier (persistent kernel) ----------------
__device__ inline void grid_barrier(int* bar) {
  __syncthreads();
  if (threadIdx.x == 0) {
    int gen = __hip_atomic_load(&bar[1], __ATOMIC_RELAXED, __HIP_MEMORY_SCOPE_AGENT);
    int arrived = __hip_atomic_fetch_add(&bar[0], 1, __ATOMIC_ACQ_REL,
                                         __HIP_MEMORY_SCOPE_AGENT);
    if (arrived == NBLK - 1) {
      __hip_atomic_store(&bar[0], 0, __ATOMIC_RELAXED, __HIP_MEMORY_SCOPE_AGENT);
      __hip_atomic_fetch_add(&bar[1], 1, __ATOMIC_RELEASE, __HIP_MEMORY_SCOPE_AGENT);
    } else {
      while (__hip_atomic_load(&bar[1], __ATOMIC_ACQUIRE,
                               __HIP_MEMORY_SCOPE_AGENT) == gen)
        __builtin_amdgcn_s_sleep(2);
    }
  }
  __syncthreads();
}

// ---------------- kernel 3: persistent recurrence ----------------
// 32 blocks x 256 threads = 256 waves; wave owns a fixed 16(batch) x 16(hidden)
// patch for all 512 steps; cell state c stays in registers. K-loop pipelined
// with a 2x unroll and alternating register sets (no rotation copies).
__global__ __launch_bounds__(256) void lstm_steps(const float* __restrict__ xproj,
                                                  const bf16_t* __restrict__ WhB,
                                                  const float* __restrict__ bh,
                                                  bf16_t* __restrict__ hbuf,
                                                  float* __restrict__ out,
                                                  int* __restrict__ bar) {
  const int wave = blockIdx.x * 8 + (threadIdx.x >> 5);   // 0..255
  const int lane = threadIdx.x & 31;
  const int mt = wave >> 6;                               // 0..3  (batch tile)
  const int ht = wave & 63;                               // 0..63 (hidden tile)
  const int b0 = mt * 16, h0 = ht * 16;
  const int colh = h0 + (lane & 15);
  const int rofs = (lane & 16) ? 8 : 0;

  bf16_t* hb0 = hbuf;
  bf16_t* hb1 = hbuf + (size_t)BATCH * DIM;

  const bf16_t* brow[4];
#pragma unroll
  for (int g = 0; g < 4; ++g) brow[g] = WhB + ((size_t)g * DIM + h0) * DIM;

  float biasv[4];
#pragma unroll
  for (int g = 0; g < 4; ++g) biasv[g] = bh[g * DIM + colh];

  // h(-1) = 0 : each wave zeroes its own patch of buffer 0
#pragma unroll
  for (int r = 0; r < 8; ++r)
    hb0[(size_t)(b0 + r + rofs) * DIM + colh] = (__bf16)0.0f;

  float cst[8];
#pragma unroll
  for (int r = 0; r < 8; ++r) cst[r] = 0.0f;

  grid_barrier(bar);

  for (int t = 0; t < T_SEQ; ++t) {
    const bf16_t* hin = (t & 1) ? hb1 : hb0;
    bf16_t* hout      = (t & 1) ? hb0 : hb1;

    // accumulators = xproj_t + bh (C-fragment layout: all 8 elems share column)
    v8f acc[4];
#pragma unroll
    for (int g = 0; g < 4; ++g) {
      const float* xp = xproj + (((size_t)t * 4 + g) * BATCH) * DIM;
#pragma unroll
      for (int r = 0; r < 8; ++r)
        acc[g][r] = xp[(size_t)(b0 + r + rofs) * DIM + colh] + biasv[g];
    }

    // gates += h @ Wh^T for this wave's 4 gate tiles (Wh slice is L2-resident)
    const bf16_t* arow = hin + (size_t)b0 * DIM;

    v16bf aA, aB, bA[4], bB[4];
    aA = load_a_bf16(arow, DIM, lane, 0);
#pragma unroll
    for (int g = 0; g < 4; ++g) bA[g] = load_b_bf16(brow[g], DIM, lane, 0);

    for (int kk = 0; kk < KSTEPS; kk += 2) {
      int k1 = (kk + 1) * 32;
      aB = load_a_bf16(arow, DIM, lane, k1);
#pragma unroll
      for (int g = 0; g < 4; ++g) bB[g] = load_b_bf16(brow[g], DIM, lane, k1);
#pragma unroll
      for (int g = 0; g < 4; ++g) acc[g] = wmma_bf16(aA, bA[g], acc[g]);

      int k2 = (kk + 2 < KSTEPS) ? (kk + 2) * 32 : 0;   // wrap: dead but in-bounds
      aA = load_a_bf16(arow, DIM, lane, k2);
#pragma unroll
      for (int g = 0; g < 4; ++g) bA[g] = load_b_bf16(brow[g], DIM, lane, k2);
#pragma unroll
      for (int g = 0; g < 4; ++g) acc[g] = wmma_bf16(aB, bB[g], acc[g]);
    }

    // pointwise LSTM cell (reference's nonstandard sigmoid on the cell update)
#pragma unroll
    for (int r = 0; r < 8; ++r) {
      float gi = sigmoidf(acc[0][r]);
      float gf = sigmoidf(acc[1][r]);
      float go = sigmoidf(acc[2][r]);
      float gc = tanhf(acc[3][r]);
      float cn = sigmoidf(gf * cst[r] + gi * gc);
      float hn = tanhf(cn) * go;
      cst[r] = cn;
      size_t oi = (size_t)(b0 + r + rofs) * DIM + colh;
      hout[oi] = (__bf16)hn;
      if (t == T_SEQ - 1) out[oi] = hn;
    }
    grid_barrier(bar);   // publish h(t) before anyone reads it at t+1
  }
}

// ---------------- host launcher ----------------
extern "C" void kernel_launch(void* const* d_in, const int* in_sizes, int n_in,
                              void* d_out, int out_size, void* d_ws, size_t ws_size,
                              hipStream_t stream) {
  const float* x  = (const float*)d_in[0];
  const float* Wx = (const float*)d_in[1];
  const float* bx = (const float*)d_in[2];
  const float* Wh = (const float*)d_in[3];
  const float* bh = (const float*)d_in[4];
  float* out = (float*)d_out;

  char* ws = (char*)d_ws;
  int*    bar   = (int*)(ws + BAR_OFF);
  bf16_t* WxB   = (bf16_t*)(ws + WXB_OFF);
  bf16_t* WhB   = (bf16_t*)(ws + WHB_OFF);
  bf16_t* hbuf  = (bf16_t*)(ws + HBUF_OFF);
  float*  xproj = (float*)(ws + XPROJ_OFF);

  {
    const size_t nelem = 2ull * NTOT * DIM;            // both weight tensors
    int blocks = (int)(nelem / (256ull * 4ull));       // 8192
    cvt_weights<<<blocks, 256, 0, stream>>>(Wx, Wh, WxB, WhB, bar);
  }
  {
    dim3 grid(T_SEQ * BATCH / 16, (NTOT / 16) / (4 * 8));  // 2048 x 8
    xproj_gemm<<<grid, 256, 0, stream>>>(x, WxB, bx, xproj);
  }
  lstm_steps<<<NBLK, 256, 0, stream>>>(xproj, WhB, bh, hbuf, out, bar);
}
